// SparseConv2d_3607772529341
// MI455X (gfx1250) — matrix operand order repaired and checked by reading, not compiled, not run on hardware
//
#include <hip/hip_runtime.h>

typedef __attribute__((ext_vector_type(16))) __bf16 v16bf;
typedef __attribute__((ext_vector_type(8)))  __bf16 v8bf;
typedef __attribute__((ext_vector_type(8)))  float  v8f;

#define CIN  256
#define COUT 256
#define HH   64
#define WW   64
#define NIMG 32

#define X_ELEMS (NIMG * CIN * HH * WW)   // 33,554,432 bf16 in ws
#define KSTEPS  72                       // 9 taps * 8 ci-chunks of 32

// round-to-nearest-even fp32 -> bf16 bit pattern
static __device__ __forceinline__ unsigned short f2bf(float f) {
    union { float f; unsigned u; } c; c.f = f;
    unsigned u = c.u;
    return (unsigned short)((u + 0x7FFFu + ((u >> 16) & 1u)) >> 16);
}

// ---------------- pre-pass 1: x NCHW fp32 -> NHWC bf16 ----------------
__global__ __launch_bounds__(256) void x_to_nhwc_bf16(
    const float* __restrict__ x, unsigned short* __restrict__ xt) {
    __shared__ unsigned short tile[64 * 40];   // [w][ci], padded stride
    int b  = blockIdx.x;        // ((n*64 + h) * 8) + cb
    int cb = b & 7;             // 32-ci block
    int nh = b >> 3;
    int h = nh & 63, n = nh >> 6;
    int w  = threadIdx.x & 63;
    int c4 = threadIdx.x >> 6;  // 0..3
#pragma unroll
    for (int p = 0; p < 8; ++p) {
        int cil = p * 4 + c4;
        float v = x[(((n << 8) + (cb << 5) + cil) << 12) + (h << 6) + w];
        tile[w * 40 + cil] = f2bf(v);
    }
    __syncthreads();
    int seg = threadIdx.x & 3, w2 = threadIdx.x >> 2;
    uint4 v = *(const uint4*)&tile[w2 * 40 + (seg << 3)];
    *(uint4*)&xt[(((nh << 6) + w2) << 8) + (cb << 5) + (seg << 3)] = v;
}

// ---------------- pre-pass 2: weight OIHW fp32 -> [co][tap][ci] bf16 ----------------
__global__ __launch_bounds__(256) void weight_reorder_bf16(
    const float* __restrict__ w, unsigned short* __restrict__ wt) {
    int i = blockIdx.x * 256 + threadIdx.x;
    if (i >= COUT * 9 * CIN) return;
    int ci  = i & 255;
    int tap = (i >> 8) % 9;
    int co  = i / (9 * 256);
    wt[i] = f2bf(w[(co * CIN + ci) * 9 + tap]);
}

static __device__ __forceinline__ void tile_coords(int kt, int& t, int& c0,
                                                   int& dh, int& dw) {
    t  = kt >> 3;
    c0 = (kt & 7) << 5;
    int d3 = t / 3;
    dh = d3 - 1;
    dw = (t - d3 * 3) - 1;
}

// ---------------- main: implicit-GEMM conv, bf16 WMMA ----------------
// D[co][m]: block tile 128 co x 128 spatial; 8 wave32, each 32co x 64sp (2x4 frags)
__global__ __launch_bounds__(256) void conv3x3_wmma(
    const unsigned short* __restrict__ xt, const unsigned short* __restrict__ wt,
    const float* __restrict__ bias, float* __restrict__ out) {

    __shared__ unsigned short lw[2][128 * 40];  // weight tile [co][k], stride 40
    __shared__ unsigned short lx[2][128 * 40];  // im2col tile [m][k], stride 40

    const int tid  = threadIdx.x;
    const int lane = tid & 31;
    const int wv   = tid >> 5;
    const int wm   = wv & 3;
    const int wn   = wv >> 2;
    const int hi16 = lane >> 4;
    const int l16  = lane & 15;

    const int spatBase = blockIdx.x * 128;
    const int coBase   = blockIdx.y * 128;
    const int n  = spatBase >> 12;
    const int h0 = (spatBase & 4095) >> 6;

    // staging geometry: each thread owns row (tid>>1), segments q0,q0+1 (16B each)
    const int srow = tid >> 1;
    const int q0   = (tid & 1) << 1;
    const int sw   = srow & 63;          // w of this staging row
    const int sh   = srow >> 6;          // 0/1

    v8f acc[2][4];
#pragma unroll
    for (int i = 0; i < 2; ++i)
#pragma unroll
        for (int j = 0; j < 4; ++j)
            acc[i][j] = (v8f){0.f,0.f,0.f,0.f,0.f,0.f,0.f,0.f};

    union F16 { v16bf v; v8bf h[2]; };

    // ---- issue staging for tile kt into buffer b: weights async->LDS, x->regs
    uint4 xr0, xr1;
    auto issue_stage = [&](int kt, int b) {
        int t, c0, dh, dw;
        tile_coords(kt, t, c0, dh, dw);
        // weights: 2x 16B async direct to LDS (always in-bounds)
        {
            int co = coBase + srow;
            unsigned long long ga = (unsigned long long)(size_t)
                (wt + (((co * 9 + t) << 8) + c0 + (q0 << 3)));
            unsigned lds0 = (unsigned)(size_t)&lw[b][srow * 40 + (q0 << 3)];
            asm volatile("global_load_async_to_lds_b128 %0, %1, off"
                         :: "v"(lds0), "v"(ga) : "memory");
            asm volatile("global_load_async_to_lds_b128 %0, %1, off offset:16"
                         :: "v"(lds0), "v"(ga) : "memory");
        }
        // x: 2x 16B global -> regs, zero-select for padding (no EXEC divergence)
        {
            int w_ = sw + dw;
            int h_ = h0 + sh + dh;
            bool inb = ((unsigned)h_ < HH) && ((unsigned)w_ < WW);
            int hc = h_ < 0 ? 0 : (h_ > 63 ? 63 : h_);
            int wc = w_ < 0 ? 0 : (w_ > 63 ? 63 : w_);
            const uint4* g = (const uint4*)
                (xt + (((((n << 6) + hc) << 6) + wc) << 8) + c0 + (q0 << 3));
            uint4 a = g[0], c = g[1];
            xr0.x = inb ? a.x : 0u; xr0.y = inb ? a.y : 0u;
            xr0.z = inb ? a.z : 0u; xr0.w = inb ? a.w : 0u;
            xr1.x = inb ? c.x : 0u; xr1.y = inb ? c.y : 0u;
            xr1.z = inb ? c.z : 0u; xr1.w = inb ? c.w : 0u;
        }
    };

    issue_stage(0, 0);

    for (int kt = 0; kt < KSTEPS; ++kt) {
        const int cur = kt & 1;

        // complete x staging into lx[cur]
        *(uint4*)&lx[cur][srow * 40 + (q0 << 3)]      = xr0;
        *(uint4*)&lx[cur][srow * 40 + (q0 << 3) + 8]  = xr1;
        // weight asyncs for this tile must be in LDS before the barrier
        asm volatile("s_wait_asynccnt 0" ::: "memory");
        __syncthreads();

        // overlap next tile's global traffic with this tile's WMMAs
        if (kt + 1 < KSTEPS) issue_stage(kt + 1, cur ^ 1);

        // ---- fragments (ISA 7.12.2 layouts) ----
        F16 af[2], bfr[4];
#pragma unroll
        for (int mf = 0; mf < 2; ++mf) {
            int row = wm * 32 + mf * 16 + l16;
            const v8bf* p = (const v8bf*)&lw[cur][row * 40];
            af[mf].h[0] = p[hi16];        // A: K 0-7 / 8-15
            af[mf].h[1] = p[2 + hi16];    //    K 16-23 / 24-31
        }
#pragma unroll
        for (int nf = 0; nf < 4; ++nf) {
            int row = wn * 64 + nf * 16 + l16;
            const v8bf* p = (const v8bf*)&lx[cur][row * 40];
            bfr[nf].h[0] = p[hi16];           // B: K 0-7 / 8-15
            bfr[nf].h[1] = p[2 + hi16];       //    K 16-23 / 24-31
        }
#pragma unroll
        for (int mf = 0; mf < 2; ++mf)
#pragma unroll
            for (int nf = 0; nf < 4; ++nf)
                acc[mf][nf] = __builtin_amdgcn_wmma_f32_16x16x32_bf16(
                    false, af[mf].v, false, bfr[nf].v,
                    (short)0, acc[mf][nf], false, false);
    }

    // ---- epilogue: +bias, fp32 NCHW stores (half-wave = 16 consecutive w) ----
#pragma unroll
    for (int mf = 0; mf < 2; ++mf) {
#pragma unroll
        for (int r = 0; r < 8; ++r) {
            int co = coBase + wm * 32 + mf * 16 + r + hi16 * 8;
            float bv = bias[co];
#pragma unroll
            for (int nf = 0; nf < 4; ++nf) {
                int m = spatBase + wn * 64 + nf * 16 + l16;
                out[(((n << 8) + co) << 12) + (m & 4095)] = acc[mf][nf][r] + bv;
            }
        }
    }
}

extern "C" void kernel_launch(void* const* d_in, const int* in_sizes, int n_in,
                              void* d_out, int out_size, void* d_ws, size_t ws_size,
                              hipStream_t stream) {
    const float* x  = (const float*)d_in[0];
    const float* w  = (const float*)d_in[1];
    const float* b  = (const float*)d_in[2];
    float* out = (float*)d_out;

    unsigned short* xt = (unsigned short*)d_ws;            // 67.1 MB NHWC bf16
    unsigned short* wt = xt + X_ELEMS;                     // 1.18 MB [co][tap][ci]

    x_to_nhwc_bf16<<<NIMG * HH * (CIN / 32), 256, 0, stream>>>(x, xt);
    weight_reorder_bf16<<<(COUT * 9 * CIN + 255) / 256, 256, 0, stream>>>(w, wt);

    dim3 grid((NIMG * HH * WW) / 128, COUT / 128);         // 1024 x 2
    conv3x3_wmma<<<grid, 256, 0, stream>>>(xt, wt, b, out);
}
